// CoordinateSetAttention_67095979099218
// MI455X (gfx1250) — compile-verified, matrix operand order checked by reading, not run on hardware
//
#include <hip/hip_runtime.h>
#include <cstdint>
#include <cstddef>

// ---------------------------------------------------------------------------
// Types / WMMA helpers (gfx1250, wave32, v_wmma_f32_16x16x32_f16)
// ---------------------------------------------------------------------------
typedef _Float16 h8   __attribute__((ext_vector_type(8)));
typedef _Float16 v16h __attribute__((ext_vector_type(16)));
typedef float    v8f  __attribute__((ext_vector_type(8)));

__device__ __forceinline__ v8f vzero() {
  v8f z;
#pragma unroll
  for (int i = 0; i < 8; ++i) z[i] = 0.0f;
  return z;
}

__device__ __forceinline__ v16h mk16(h8 lo, h8 hi) {
  v16h r;
#pragma unroll
  for (int i = 0; i < 8; ++i) { r[i] = lo[i]; r[8 + i] = hi[i]; }
  return r;
}

// Fragment loader for v_wmma_f32_16x16x32_f16.
//  A-operand: base is row-major [M][K], idx0 = row tile origin  (lane&15 = row)
//  B-operand: base is [N][K] row-major,  idx0 = col tile origin  (lane&15 = col)
// Per ISA layout: lanes 0-15 hold K = {0..7, 16..23}, lanes 16-31 K = {8..15, 24..31},
// i.e. two contiguous 8-half (16B) chunks per lane -> two b128 loads.
__device__ __forceinline__ v16h load_frag16(const _Float16* base, int ld,
                                            int idx0, int k0, int lane) {
  int half = lane >> 4, rc = lane & 15;
  const _Float16* p = base + (size_t)(idx0 + rc) * ld + k0 + half * 8;
  h8 lo = *(const h8*)p;
  h8 hi = *(const h8*)(p + 16);
  return mk16(lo, hi);
}

__device__ __forceinline__ v8f wmma16(v16h a, v16h b, v8f c) {
  return __builtin_amdgcn_wmma_f32_16x16x32_f16(false, a, false, b,
                                                (short)0, c, false, false);
}

// ---------------------------------------------------------------------------
// Constants / workspace layout (bytes, all offsets multiple of 256)
// ---------------------------------------------------------------------------
constexpr size_t OFF_WQ      = 0;            // 128x128 f16
constexpr size_t OFF_WK      = 32768;        // 128x128 f16
constexpr size_t OFF_WO      = 65536;        // 64x128  f16
constexpr size_t OFF_SIN80   = 81920;        // 80x64   f32
constexpr size_t OFF_COS80   = 102400;
constexpr size_t OFF_SIN160  = 122880;       // 160x64  f32
constexpr size_t OFF_COS160  = 163840;
constexpr size_t OFF_VT80    = 204800;       // 128x96  f16 (zero padded)
constexpr size_t OFF_VT160   = 229376;       // 128x160 f16
constexpr size_t OFF_MASK    = 270336;       // 320x160 f32 (reused per diag)
constexpr size_t OFF_MAT     = 475136;       // up to 51200x128 f16 (reused)
constexpr size_t OFF_Q       = 13582336;     // up to 51200x128 f16
constexpr size_t OFF_K       = 26689536;
constexpr size_t OFF_O       = 39796736;
constexpr size_t OFF_RESROW  = 52903936;     // 12800x64 f32
constexpr size_t OFF_RESCOL  = 56180736;
constexpr size_t OFF_RESDIAG = 59457536;     // 51200x64 f32 (reused)
constexpr size_t OFF_DCROPP  = 72564736;     // 2x80x80x64 f32
constexpr size_t OFF_DCROPM  = 75841536;     // end: 79118336

constexpr float LOG1E4_64 = 0.14391156831212787f; // ln(10000)/64

// ---------------------------------------------------------------------------
// Prep kernels
// ---------------------------------------------------------------------------
__global__ void k_convert(const float* __restrict__ Wi, const float* __restrict__ Wout,
                          _Float16* __restrict__ wq, _Float16* __restrict__ wk,
                          _Float16* __restrict__ wo) {
  int idx = blockIdx.x * 256 + threadIdx.x;
  if (idx < 16384) { wq[idx] = (_Float16)Wi[idx]; wk[idx] = (_Float16)Wi[16384 + idx]; }
  if (idx < 8192)  wo[idx] = (_Float16)Wout[idx];
}

__global__ void k_tables(float* __restrict__ s80, float* __restrict__ c80,
                         float* __restrict__ s160, float* __restrict__ c160) {
  int idx = blockIdx.x * 256 + threadIdx.x;
  if (idx < 80 * 64) {
    int l = idx / 64, i = idx % 64;
    float X = (float)(l + 1) * expf(-(float)i * LOG1E4_64);
    s80[idx] = sinf(X); c80[idx] = cosf(X);
  } else if (idx < 80 * 64 + 160 * 64) {
    int j = idx - 80 * 64;
    int l = j / 64, i = j % 64;
    float X = (float)(l + 1) * expf(-(float)i * LOG1E4_64);
    s160[j] = sinf(X); c160[j] = cosf(X);
  }
}

// v = P @ Wv^T + bv, stored transposed vT[e][l] (f16), K-padded with zeros to Lp.
__global__ void k_vproj(const float* __restrict__ Wi, const float* __restrict__ bi,
                        _Float16* __restrict__ vT, int L, int Lp) {
  int idx = blockIdx.x * 256 + threadIdx.x;
  if (idx >= 128 * Lp) return;
  int l = idx % Lp, e = idx / Lp;
  if (l >= L) { vT[idx] = (_Float16)0.0f; return; }
  const float* Wv = Wi + 256 * 128 + (size_t)e * 128;
  float acc = bi[256 + e];
  float lp1 = (float)(l + 1);
#pragma unroll 8
  for (int i = 0; i < 64; ++i) {
    float X = lp1 * expf(-(float)i * LOG1E4_64);
    acc += sinf(X) * Wv[2 * i] + cosf(X) * Wv[2 * i + 1];
  }
  vT[idx] = (_Float16)acc;
}

// ---------------------------------------------------------------------------
// Sequence builders (f32 -> f16)
// ---------------------------------------------------------------------------
__global__ void k_build_rowcol(const float* __restrict__ x, _Float16* __restrict__ mat,
                               int colmode) {
  int idx = blockIdx.x * 256 + threadIdx.x;
  if (idx >= 12800 * 128) return;
  int c = idx & 127, t = idx >> 7;
  int Lidx = t % 80, A = (t / 80) % 80, b = t / 6400;
  float v = colmode ? x[(((size_t)b * 128 + c) * 80 + Lidx) * 80 + A]
                    : x[(((size_t)b * 128 + c) * 80 + A) * 80 + Lidx];
  mat[(size_t)t * 128 + c] = (_Float16)v;
}

__global__ void k_build_diag(const float* __restrict__ x, _Float16* __restrict__ mat,
                             float si) {
  const float co = 0.70710678118654752f;
  int idx = blockIdx.x * 256 + threadIdx.x;
  if (idx >= 51200 * 128) return;
  int c = idx & 127, t = idx >> 7;
  int xx = t % 160, yy = (t / 160) % 160, b = t / 25600;
  float dx = xx - 79.5f, dy = yy - 79.5f;
  float sx = co * dx + si * dy + 79.5f;
  float sy = -si * dx + co * dy + 79.5f;
  float x0 = floorf(sx), y0 = floorf(sy);
  float wx = sx - x0, wy = sy - y0;
  int xi = (int)x0, yi = (int)y0;
  auto samp = [&](int py, int px) -> float {
    if (py < 40 || py >= 120 || px < 40 || px >= 120) return 0.0f;
    return x[(((size_t)b * 128 + c) * 80 + (py - 40)) * 80 + (px - 40)];
  };
  float v = samp(yi, xi) * (1.f - wy) * (1.f - wx)
          + samp(yi, xi + 1) * (1.f - wy) * wx
          + samp(yi + 1, xi) * wy * (1.f - wx)
          + samp(yi + 1, xi + 1) * wy * wx;
  mat[(size_t)t * 128 + c] = (_Float16)v;
}

__global__ void k_build_mask(float* __restrict__ mask, float si) {
  const float co = 0.70710678118654752f;
  int idx = blockIdx.x * 256 + threadIdx.x;
  if (idx >= 320 * 160) return;
  int xx = idx % 160, yy = (idx / 160) % 160;
  float dx = xx - 79.5f, dy = yy - 79.5f;
  float sx = co * dx + si * dy + 79.5f;
  float sy = -si * dx + co * dy + 79.5f;
  int rx = (int)rintf(sx), ry = (int)rintf(sy);
  mask[idx] = (rx >= 40 && rx < 120 && ry >= 40 && ry < 120) ? 1.0f : 0.0f;
}

// ---------------------------------------------------------------------------
// Projection GEMM: Out(f16)[M,128] = A(f16)[M,128] @ W[N=128,K=128]^T + bias
// Block = 256 thr (8 waves) -> M-tile 128, full N=128; weights cached in LDS.
// ---------------------------------------------------------------------------
__global__ void __launch_bounds__(256)
k_gemm_proj(const _Float16* __restrict__ A, const _Float16* __restrict__ W,
            const float* __restrict__ bias, _Float16* __restrict__ Out) {
  __shared__ __align__(16) _Float16 sW[128 * 128];
  int tid = threadIdx.x;
  for (int i = tid; i < 2048; i += 256) ((h8*)sW)[i] = ((const h8*)W)[i];
  __syncthreads();
  int wave = tid >> 5, lane = tid & 31;
  int m0 = blockIdx.x * 128 + wave * 16;
  v8f acc[8];
#pragma unroll
  for (int n = 0; n < 8; ++n) acc[n] = vzero();
#pragma unroll
  for (int ks = 0; ks < 128; ks += 32) {
    v16h a = load_frag16(A, 128, m0, ks, lane);
#pragma unroll
    for (int n = 0; n < 8; ++n) {
      v16h b = load_frag16(sW, 128, n * 16, ks, lane);
      acc[n] = wmma16(a, b, acc[n]);
    }
  }
  int col16 = lane & 15, rowg = (lane >> 4) * 8;
#pragma unroll
  for (int n = 0; n < 8; ++n) {
    int col = n * 16 + col16;
    float bv = bias[col];
#pragma unroll
    for (int r = 0; r < 8; ++r)
      Out[(size_t)(m0 + rowg + r) * 128 + col] = (_Float16)(acc[n][r] + bv);
  }
}

// ---------------------------------------------------------------------------
// Fused per-(sequence, head) attention: logits -> exact softmax -> attn @ v.
// Whole L x L logits live in C-fragments; softmax via 16-lane shfl_xor
// butterflies; C-layout -> A-layout via per-wave LDS tile (no block barrier,
// same-wave DS ordering + s_wait_dscnt).
// ---------------------------------------------------------------------------
template <int L, int Lp>
__global__ void __launch_bounds__(256)
k_attn(const _Float16* __restrict__ Q, const _Float16* __restrict__ Km,
       const _Float16* __restrict__ vT, const float* __restrict__ mask,
       _Float16* __restrict__ O) {
  constexpr int NT = L / 16;
  constexpr int AST = 176;  // LDS attn row stride (halfs), 16B-aligned
  __shared__ __align__(16) _Float16 sAttn[8][16 * AST];
  int seq = blockIdx.x, head = blockIdx.y;
  int tid = threadIdx.x, wave = tid >> 5, lane = tid & 31;
  int col16 = lane & 15, rowg = (lane >> 4) * 8;
  _Float16* myA = &sAttn[wave][0];
  const float scale = 0.17677669529663689f;  // 1/sqrt(32)

  for (int rt = wave; rt < NT; rt += 8) {
    if (Lp > L) {  // zero the K-padding columns
      for (int i = lane; i < (Lp - L) * 16; i += 32) {
        int rr = i % 16, cc = L + i / 16;
        myA[rr * AST + cc] = (_Float16)0.0f;
      }
    }
    int m0 = seq * L + rt * 16;
    v16h qa = load_frag16(Q, 128, m0, head * 32, lane);
    v8f cf[NT];
#pragma unroll
    for (int ct = 0; ct < NT; ++ct) {
      v16h kb = load_frag16(Km, 128, seq * L + ct * 16, head * 32, lane);
      cf[ct] = wmma16(qa, kb, vzero());
    }
    float rmax[8], rsum[8];
#pragma unroll
    for (int r = 0; r < 8; ++r) rmax[r] = -3.0e38f;
#pragma unroll
    for (int ct = 0; ct < NT; ++ct) {
      float mb = mask ? mask[seq * L + ct * 16 + col16] : 0.0f;
#pragma unroll
      for (int r = 0; r < 8; ++r) {
        float v = cf[ct][r] * scale + mb;
        cf[ct][r] = v;
        rmax[r] = fmaxf(rmax[r], v);
      }
    }
#pragma unroll
    for (int r = 0; r < 8; ++r) {
      float v = rmax[r];
      v = fmaxf(v, __shfl_xor(v, 1, 32));
      v = fmaxf(v, __shfl_xor(v, 2, 32));
      v = fmaxf(v, __shfl_xor(v, 4, 32));
      v = fmaxf(v, __shfl_xor(v, 8, 32));
      rmax[r] = v;
      rsum[r] = 0.0f;
    }
#pragma unroll
    for (int ct = 0; ct < NT; ++ct)
#pragma unroll
      for (int r = 0; r < 8; ++r) {
        float e = __expf(cf[ct][r] - rmax[r]);
        cf[ct][r] = e;
        rsum[r] += e;
      }
#pragma unroll
    for (int r = 0; r < 8; ++r) {
      float v = rsum[r];
      v += __shfl_xor(v, 1, 32);
      v += __shfl_xor(v, 2, 32);
      v += __shfl_xor(v, 4, 32);
      v += __shfl_xor(v, 8, 32);
      rsum[r] = 1.0f / v;
    }
    // C-layout -> A-layout through per-wave LDS tile (f16 attn)
#pragma unroll
    for (int ct = 0; ct < NT; ++ct)
#pragma unroll
      for (int r = 0; r < 8; ++r)
        myA[(rowg + r) * AST + ct * 16 + col16] = (_Float16)(cf[ct][r] * rsum[r]);
    asm volatile("s_wait_dscnt 0" ::: "memory");  // same-wave DS RAW fence

    v8f oc0 = vzero(), oc1 = vzero();
#pragma unroll
    for (int ks = 0; ks < Lp; ks += 32) {
      v16h aa = load_frag16(myA, AST, 0, ks, lane);
      v16h b0 = load_frag16(vT, Lp, head * 32, ks, lane);
      v16h b1 = load_frag16(vT, Lp, head * 32 + 16, ks, lane);
      oc0 = wmma16(aa, b0, oc0);
      oc1 = wmma16(aa, b1, oc1);
    }
#pragma unroll
    for (int r = 0; r < 8; ++r) {
      size_t o = (size_t)(m0 + rowg + r) * 128 + head * 32;
      O[o + col16]      = (_Float16)oc0[r];
      O[o + 16 + col16] = (_Float16)oc1[r];
    }
  }
}

// ---------------------------------------------------------------------------
// Out-projection (only 64 needed cols) + bias + to_relative rotary twist.
// Pair partner (2i <-> 2i+1) sits in the adjacent lane -> shfl_xor(1).
// ---------------------------------------------------------------------------
__global__ void __launch_bounds__(256)
k_outproj(const _Float16* __restrict__ Ofeat, const _Float16* __restrict__ Wo,
          const float* __restrict__ bo, const float* __restrict__ sinT,
          const float* __restrict__ cosT, float* __restrict__ res, int L) {
  __shared__ __align__(16) _Float16 sW[64 * 128];
  int tid = threadIdx.x;
  for (int i = tid; i < 1024; i += 256) ((h8*)sW)[i] = ((const h8*)Wo)[i];
  __syncthreads();
  int wave = tid >> 5, lane = tid & 31, col16 = lane & 15, rowg = (lane >> 4) * 8;
  int m0 = blockIdx.x * 128 + wave * 16;
  v8f acc[4];
#pragma unroll
  for (int n = 0; n < 4; ++n) acc[n] = vzero();
#pragma unroll
  for (int ks = 0; ks < 128; ks += 32) {
    v16h a = load_frag16(Ofeat, 128, m0, ks, lane);
#pragma unroll
    for (int n = 0; n < 4; ++n) {
      v16h b = load_frag16(sW, 128, n * 16, ks, lane);
      acc[n] = wmma16(a, b, acc[n]);
    }
  }
#pragma unroll
  for (int n = 0; n < 4; ++n) {
    int col = n * 16 + col16;
    float bv = bo[col];
    int i2 = col >> 1;
    bool even = (col & 1) == 0;
#pragma unroll
    for (int r = 0; r < 8; ++r) {
      int row = m0 + rowg + r;
      int l = row % L;
      float v = acc[n][r] + bv;
      float p = __shfl_xor(v, 1, 32);
      float sn = sinT[l * 64 + i2], cs = cosT[l * 64 + i2];
      res[(size_t)row * 64 + col] = even ? (v * cs - p * sn) : (v * cs + p * sn);
    }
  }
}

// ---------------------------------------------------------------------------
// Back-rotate diag result by -angle, crop center 80x80.
// ---------------------------------------------------------------------------
__global__ void k_backrot(const float* __restrict__ resD, float* __restrict__ dcrop,
                          float si) {
  const float co = 0.70710678118654752f;
  int idx = blockIdx.x * 256 + threadIdx.x;
  if (idx >= 2 * 80 * 80 * 64) return;
  int c = idx & 63, t = idx >> 6;
  int X = t % 80, Y = (t / 80) % 80, b = t / 6400;
  float dx = (40 + X) - 79.5f, dy = (40 + Y) - 79.5f;
  float sx = co * dx - si * dy + 79.5f;   // rotate by -theta
  float sy = si * dx + co * dy + 79.5f;
  float x0 = floorf(sx), y0 = floorf(sy);
  float wx = sx - x0, wy = sy - y0;
  int xi = (int)x0, yi = (int)y0;
  auto samp = [&](int py, int px) -> float {
    if (py < 0 || py >= 160 || px < 0 || px >= 160) return 0.0f;
    return resD[(((size_t)b * 160 + py) * 160 + px) * 64 + c];
  };
  dcrop[idx] = samp(yi, xi) * (1.f - wy) * (1.f - wx)
             + samp(yi, xi + 1) * (1.f - wy) * wx
             + samp(yi + 1, xi) * wy * (1.f - wx)
             + samp(yi + 1, xi + 1) * wy * wx;
}

// ---------------------------------------------------------------------------
// Final: gather 4 sources, 5-compare sorting network, concat with x.
// ---------------------------------------------------------------------------
__global__ void k_final(const float* __restrict__ x, const float* __restrict__ rRow,
                        const float* __restrict__ rCol, const float* __restrict__ dP,
                        const float* __restrict__ dM, float* __restrict__ out) {
  int idx = blockIdx.x * 256 + threadIdx.x;
  if (idx >= 2 * 384 * 6400) return;
  int xx = idx % 80, y = (idx / 80) % 80, ch = (idx / 6400) % 384, b = idx / (6400 * 384);
  if (ch < 128) { out[idx] = x[(((size_t)b * 128 + ch) * 80 + y) * 80 + xx]; return; }
  int s = (ch - 128) >> 6, c = (ch - 128) & 63;
  size_t p = (((size_t)b * 80 + y) * 80 + xx) * 64 + c;
  float a0 = rRow[p];
  float a1 = rCol[(((size_t)b * 80 + xx) * 80 + y) * 64 + c];
  float a2 = dP[p];
  float a3 = dM[p];
  float t;
  if (a0 > a1) { t = a0; a0 = a1; a1 = t; }
  if (a2 > a3) { t = a2; a2 = a3; a3 = t; }
  if (a0 > a2) { t = a0; a0 = a2; a2 = t; }
  if (a1 > a3) { t = a1; a1 = a3; a3 = t; }
  if (a1 > a2) { t = a1; a1 = a2; a2 = t; }
  out[idx] = (s == 0) ? a0 : (s == 1) ? a1 : (s == 2) ? a2 : a3;
}

// ---------------------------------------------------------------------------
// Host driver
// ---------------------------------------------------------------------------
extern "C" void kernel_launch(void* const* d_in, const int* in_sizes, int n_in,
                              void* d_out, int out_size, void* d_ws, size_t ws_size,
                              hipStream_t stream) {
  (void)in_sizes; (void)n_in; (void)out_size; (void)ws_size;
  const float* x    = (const float*)d_in[0];
  const float* Wi   = (const float*)d_in[1];
  const float* bi   = (const float*)d_in[2];
  const float* Wout = (const float*)d_in[3];
  const float* bo   = (const float*)d_in[4];

  char* w = (char*)d_ws;
  _Float16* wq     = (_Float16*)(w + OFF_WQ);
  _Float16* wk     = (_Float16*)(w + OFF_WK);
  _Float16* wo     = (_Float16*)(w + OFF_WO);
  float*    s80    = (float*)(w + OFF_SIN80);
  float*    c80    = (float*)(w + OFF_COS80);
  float*    s160   = (float*)(w + OFF_SIN160);
  float*    c160   = (float*)(w + OFF_COS160);
  _Float16* vt80   = (_Float16*)(w + OFF_VT80);
  _Float16* vt160  = (_Float16*)(w + OFF_VT160);
  float*    maskb  = (float*)(w + OFF_MASK);
  _Float16* matb   = (_Float16*)(w + OFF_MAT);
  _Float16* Qb     = (_Float16*)(w + OFF_Q);
  _Float16* Kb     = (_Float16*)(w + OFF_K);
  _Float16* Ob     = (_Float16*)(w + OFF_O);
  float*    resRow = (float*)(w + OFF_RESROW);
  float*    resCol = (float*)(w + OFF_RESCOL);
  float*    resDia = (float*)(w + OFF_RESDIAG);
  float*    dcropP = (float*)(w + OFF_DCROPP);
  float*    dcropM = (float*)(w + OFF_DCROPM);

  const float SQ = 0.70710678118654752f;

  k_convert<<<64, 256, 0, stream>>>(Wi, Wout, wq, wk, wo);
  k_tables<<<60, 256, 0, stream>>>(s80, c80, s160, c160);
  k_vproj<<<48, 256, 0, stream>>>(Wi, bi, vt80, 80, 96);
  k_vproj<<<80, 256, 0, stream>>>(Wi, bi, vt160, 160, 160);

  auto run = [&](int seqs, int L, const float* mask, const _Float16* vt,
                 const float* sT, const float* cT, float* res) {
    int M = seqs * L;
    k_gemm_proj<<<M / 128, 256, 0, stream>>>(matb, wq, bi, Qb);
    k_gemm_proj<<<M / 128, 256, 0, stream>>>(matb, wk, bi + 128, Kb);
    if (L == 80)
      k_attn<80, 96><<<dim3(seqs, 4), 256, 0, stream>>>(Qb, Kb, vt, mask, Ob);
    else
      k_attn<160, 160><<<dim3(seqs, 4), 256, 0, stream>>>(Qb, Kb, vt, mask, Ob);
    k_outproj<<<M / 128, 256, 0, stream>>>(Ob, wo, bo, sT, cT, res, L);
  };

  // Row attention
  k_build_rowcol<<<6400, 256, 0, stream>>>(x, matb, 0);
  run(160, 80, nullptr, vt80, s80, c80, resRow);
  // Column attention
  k_build_rowcol<<<6400, 256, 0, stream>>>(x, matb, 1);
  run(160, 80, nullptr, vt80, s80, c80, resCol);
  // Diagonal +45
  k_build_diag<<<25600, 256, 0, stream>>>(x, matb, SQ);
  k_build_mask<<<200, 256, 0, stream>>>(maskb, SQ);
  run(320, 160, maskb, vt160, s160, c160, resDia);
  k_backrot<<<3200, 256, 0, stream>>>(resDia, dcropP, SQ);
  // Diagonal -45
  k_build_diag<<<25600, 256, 0, stream>>>(x, matb, -SQ);
  k_build_mask<<<200, 256, 0, stream>>>(maskb, -SQ);
  run(320, 160, maskb, vt160, s160, c160, resDia);
  k_backrot<<<3200, 256, 0, stream>>>(resDia, dcropM, -SQ);
  // Sort + concat
  k_final<<<19200, 256, 0, stream>>>(x, resRow, resCol, dcropP, dcropM, (float*)d_out);
}
